// SharedLoRADiffusionModel_28321014350228
// MI455X (gfx1250) — matrix operand order repaired and checked by reading, not compiled, not run on hardware
//
#include <hip/hip_runtime.h>

// ---------------------------------------------------------------------------
// SharedLoRADiffusionModel forward, CDNA5 (gfx1250) WMMA implementation.
//
//   1) build_h0    : assemble 649-wide features -> bf16, K padded to 672
//   2) per layer:
//        prep_weff : fold rank-8 LoRA into weights, bf16, K-contiguous per col
//        lora_gemm : per-agent GEMM, D = Weff_tile x h_tile^T via
//                    v_wmma_f32_16x16x32_bf16 (D rows = features -> vectorized
//                    b128 epilogue stores). Tokens staged in LDS via
//                    global_load_async_to_lds_b128; weights stream from the
//                    L2-resident W_eff.
//   Final layer writes f32 into d_out with the own/neigh split.
// ---------------------------------------------------------------------------

typedef unsigned short u16;
typedef __attribute__((ext_vector_type(16))) __bf16    v16bf;
typedef __attribute__((ext_vector_type(8)))  float     v8f;
typedef __attribute__((ext_vector_type(4)))  float     f32x4;
typedef __attribute__((ext_vector_type(4)))  unsigned  u32x4;
typedef int v4i __attribute__((vector_size(16)));
typedef __attribute__((address_space(1))) v4i* gv4p;   // global int4*
typedef __attribute__((address_space(3))) v4i* lv4p;   // LDS int4*

union Frag16 { v16bf v; u32x4 q[2]; };

#define NB   2048          // batch
#define NA   64            // agents
#define NP   128           // patch dim
#define NT   (NB * NA)     // tokens
#define LD0  672           // 649 padded to multiple of 32

#if __has_builtin(__builtin_amdgcn_global_load_async_to_lds_b128)
#define USE_ASYNC_LDS 1
#else
#define USE_ASYNC_LDS 0
#endif

__device__ __forceinline__ u16 f2bf(float f) {
    union { float f; unsigned u; } x; x.f = f;
    unsigned r = x.u + 0x7FFFu + ((x.u >> 16) & 1u);   // round-to-nearest-even
    return (u16)(r >> 16);
}

__device__ __forceinline__ void copy16_to_lds(u16* lp, const u16* gp) {
#if USE_ASYNC_LDS
    __builtin_amdgcn_global_load_async_to_lds_b128(
        (gv4p)(u16*)gp, (lv4p)lp, 0, 0);
#else
    *(u32x4*)lp = *(const u32x4*)gp;
#endif
}

__device__ __forceinline__ void wait_lds_copies() {
#if USE_ASYNC_LDS
#if __has_builtin(__builtin_amdgcn_s_wait_asynccnt)
    __builtin_amdgcn_s_wait_asynccnt(0);
#else
    asm volatile("s_wait_asynccnt 0x0" ::: "memory");
#endif
#endif
}

// ---------------------------------------------------------------------------
// 1) Build h0: [noisy(128) | neigh(4*128) | presence(4) | t_feat(5) | pad->672]
// ---------------------------------------------------------------------------
__global__ void build_h0(const float* __restrict__ noisy,
                         const float* __restrict__ mask,
                         const int*   __restrict__ tsteps,
                         const int*   __restrict__ nidx,
                         const int*   __restrict__ num_steps_p,
                         u16*         __restrict__ h0)
{
    const int t = blockIdx.x;           // token
    const int b = t / NA;
    const int a = t % NA;
    const int ns  = num_steps_p[0];
    const float tau = (float)tsteps[b] / (float)(ns > 1 ? ns : 1);
    const float PI  = 3.14159265358979323846f;

    for (int k = threadIdx.x; k < LD0; k += blockDim.x) {
        float v;
        if (k < NP) {
            v = noisy[(size_t)t * NP + k];
        } else if (k < NP + 4 * NP) {
            int n = (k - NP) >> 7;
            int p = (k - NP) & 127;
            int ia = nidx[a * 4 + n]; ia = ia < 0 ? 0 : ia;
            v = noisy[((size_t)b * NA + ia) * NP + p] * mask[a * 4 + n];
        } else if (k < NP + 4 * NP + 4) {
            v = mask[a * 4 + (k - (NP + 4 * NP))];
        } else if (k < NP + 4 * NP + 9) {
            int f = k - (NP + 4 * NP + 4);
            v = (f == 0) ? tau
              : (f == 1) ? sinf(PI * tau)
              : (f == 2) ? cosf(PI * tau)
              : (f == 3) ? sinf(2.0f * PI * tau)
                         : cosf(2.0f * PI * tau);
        } else {
            v = 0.0f;                    // K padding so the WMMA K-loop is exact
        }
        h0[(size_t)t * LD0 + k] = f2bf(v);
    }
}

// ---------------------------------------------------------------------------
// 2a) Fold LoRA into the weight, store bf16 transposed: weff[a][o][k], ld = KP
// ---------------------------------------------------------------------------
__global__ void prep_weff(const float* __restrict__ W,    // (din, dout)
                          const float* __restrict__ la,   // (A, 8, din)
                          const float* __restrict__ lb,   // (A, dout, 8)
                          u16*         __restrict__ weff, // (A, dout, KP)
                          int din, int dout, int KP)
{
    const int blk = blockIdx.x;
    const int a = blk / dout;
    const int o = blk % dout;

    float lbv[8];
#pragma unroll
    for (int r = 0; r < 8; ++r) lbv[r] = lb[((size_t)a * dout + o) * 8 + r];

    const float* laA  = la + (size_t)a * 8 * din;
    u16*         orow = weff + ((size_t)a * dout + o) * KP;

    for (int k = threadIdx.x; k < KP; k += blockDim.x) {
        float v = 0.0f;
        if (k < din) {
            v = W[(size_t)k * dout + o];
#pragma unroll
            for (int r = 0; r < 8; ++r) v += 2.0f * lbv[r] * laA[(size_t)r * din + k];
        }
        orow[k] = f2bf(v);
    }
}

// ---------------------------------------------------------------------------
// 2b) WMMA GEMM, feature-major: D[n][tok] = sum_k Weff[a][n][k] * h[tok][k]
//     A-operand = 16 weight rows (K-contiguous), B-operand = 16 token rows
//     (K-contiguous, from LDS). C/D: VGPR v -> feature n = v + 8*hi (in-lane
//     contiguous) -> one b128 store per accumulator.
//     Wave tile: 32 n x 32 tok (4 acc). Block: 8 waves = 256 n x 32 tok.
// ---------------------------------------------------------------------------
template <int KP, int DOUT, bool RELU, bool FINAL>
__global__ __launch_bounds__(256) void lora_gemm(
    const u16*  __restrict__ hIn,
    const u16*  __restrict__ weff,
    const float* __restrict__ bias,
    u16*   __restrict__ hOut, int ldout,
    float* __restrict__ outF32)
{
    constexpr int STRIDE = KP + 8;           // bank skew for 16-row LDS reads
    __shared__ u16 smT[32 * STRIDE];         // 32 tokens x KP bf16

    const int tid     = threadIdx.x;
    const int a       = blockIdx.z;
    const int rowbase = blockIdx.y * 32;     // first batch index of token tile

    // ---- stage token tile: 32 rows x KP bf16, coalesced 16B chunks ----
    constexpr int CPR = KP / 8;              // 16B chunks per row
    constexpr int TOT = 32 * CPR;
    for (int idx = tid; idx < TOT; idx += 256) {
        const int row = idx / CPR;
        const int ch  = idx - row * CPR;
        const u16* gp = hIn + ((size_t)(rowbase + row) * NA + a) * KP + ch * 8;
        copy16_to_lds(smT + row * STRIDE + ch * 8, gp);
    }
    wait_lds_copies();
    __syncthreads();

    const int lane  = tid & 31;
    const int wave  = tid >> 5;              // 0..7 -> feature sub-tile
    const int nbase = blockIdx.x * 256 + wave * 32;
    if (nbase >= DOUT) return;               // wave-uniform, after the barrier

    const int c  = lane & 15;
    const int hi = lane >> 4;

    // A-operand rows (weights): lane&15 = row; K split hi*8 / +16
    const u16* wrow0 = weff + ((size_t)a * DOUT + nbase + c) * KP;
    const u16* wrow1 = wrow0 + (size_t)16 * KP;
    // B-operand rows (tokens) in LDS: lane&15 = column(token); K split hi*8 / +16
    const u16* srow0 = smT + c * STRIDE;
    const u16* srow1 = smT + (16 + c) * STRIDE;

    v8f acc[2][2] = {};                      // [n-sub][tok-sub]
    constexpr int NK = KP / 32;
#pragma unroll 2
    for (int kt = 0; kt < NK; ++kt) {
        const int ka = kt * 32 + hi * 8;     // A: 8 elems at ka, 8 at ka+16
        const int kb = kt * 32 + hi * 8;     // B: 8 elems at kb, 8 at kb+16
        Frag16 w0, w1, t0, t1;
        w0.q[0] = *(const u32x4*)(wrow0 + ka);
        w0.q[1] = *(const u32x4*)(wrow0 + ka + 16);
        w1.q[0] = *(const u32x4*)(wrow1 + ka);
        w1.q[1] = *(const u32x4*)(wrow1 + ka + 16);
        t0.q[0] = *(const u32x4*)(srow0 + kb);
        t0.q[1] = *(const u32x4*)(srow0 + kb + 16);
        t1.q[0] = *(const u32x4*)(srow1 + kb);
        t1.q[1] = *(const u32x4*)(srow1 + kb + 16);

        acc[0][0] = __builtin_amdgcn_wmma_f32_16x16x32_bf16(false, w0.v, false, t0.v, (short)0, acc[0][0], false, false);
        acc[0][1] = __builtin_amdgcn_wmma_f32_16x16x32_bf16(false, w0.v, false, t1.v, (short)0, acc[0][1], false, false);
        acc[1][0] = __builtin_amdgcn_wmma_f32_16x16x32_bf16(false, w1.v, false, t0.v, (short)0, acc[1][0], false, false);
        acc[1][1] = __builtin_amdgcn_wmma_f32_16x16x32_bf16(false, w1.v, false, t1.v, (short)0, acc[1][1], false, false);
    }

    // Epilogue: lane owns features n = nrow..nrow+7 (contiguous) for one token.
#pragma unroll
    for (int ni = 0; ni < 2; ++ni) {
        const int   nrow = nbase + ni * 16 + hi * 8;
        const f32x4 bv0  = *(const f32x4*)(bias + nrow);
        const f32x4 bv1  = *(const f32x4*)(bias + nrow + 4);
#pragma unroll
        for (int tj = 0; tj < 2; ++tj) {
            const v8f&   d    = acc[ni][tj];
            const int    tokb = rowbase + tj * 16 + c;      // batch index
            const size_t t    = (size_t)tokb * NA + a;      // token index
            float x[8];
#pragma unroll
            for (int v = 0; v < 8; ++v) {
                float y = d[v] + (v < 4 ? bv0[v] : bv1[v - 4]);
                if (RELU) y = fmaxf(y, 0.0f);
                x[v] = y;
            }
            if (!FINAL) {
                union { u32x4 q; u16 h[8]; } pk;
#pragma unroll
                for (int v = 0; v < 8; ++v) pk.h[v] = f2bf(x[v]);
                *(u32x4*)(hOut + t * ldout + nrow) = pk.q;
            } else {
                f32x4 lo = {x[0], x[1], x[2], x[3]};
                f32x4 hi4 = {x[4], x[5], x[6], x[7]};
                float* op = (nrow < NP)
                    ? outF32 + t * NP + nrow                              // own
                    : outF32 + (size_t)NT * NP + t * 512 + (nrow - NP);   // neigh
                *(f32x4*)op       = lo;
                *(f32x4*)(op + 4) = hi4;
            }
        }
    }
}

// ---------------------------------------------------------------------------
// Host launcher
// ---------------------------------------------------------------------------
extern "C" void kernel_launch(void* const* d_in, const int* in_sizes, int n_in,
                              void* d_out, int out_size, void* d_ws, size_t ws_size,
                              hipStream_t stream) {
    (void)in_sizes; (void)n_in; (void)out_size; (void)ws_size;

    const float* noisy = (const float*)d_in[0];
    const float* mask  = (const float*)d_in[1];
    const int*   tst   = (const int*)d_in[2];
    const int*   nidx  = (const int*)d_in[3];
    const int*   nstep = (const int*)d_in[4];

    const float* W0 = (const float*)d_in[5],  *b0 = (const float*)d_in[6];
    const float* la0 = (const float*)d_in[7], *lb0 = (const float*)d_in[8];
    const float* W1 = (const float*)d_in[9],  *b1 = (const float*)d_in[10];
    const float* la1 = (const float*)d_in[11], *lb1 = (const float*)d_in[12];
    const float* W2 = (const float*)d_in[13], *b2 = (const float*)d_in[14];
    const float* la2 = (const float*)d_in[15], *lb2 = (const float*)d_in[16];
    const float* W3 = (const float*)d_in[17], *b3 = (const float*)d_in[18];
    const float* la3 = (const float*)d_in[19], *lb3 = (const float*)d_in[20];

    // Workspace layout (bf16 as u16):
    u16* hA   = (u16*)d_ws;                       // T*672
    u16* hB   = hA + (size_t)NT * LD0;            // T*672
    u16* weff = hB + (size_t)NT * LD0;            // A*672*640 max

    float* out = (float*)d_out;

    build_h0<<<NT, 256, 0, stream>>>(noisy, mask, tst, nidx, nstep, hA);

    // layer 0: 649(->672) -> 512, relu        hA -> hB
    prep_weff<<<dim3(NA * 512), 256, 0, stream>>>(W0, la0, lb0, weff, 649, 512, 672);
    lora_gemm<672, 512, true, false><<<dim3(2, NB / 32, NA), 256, 0, stream>>>(
        hA, weff, b0, hB, 512, nullptr);

    // layer 1: 512 -> 512, relu               hB -> hA
    prep_weff<<<dim3(NA * 512), 256, 0, stream>>>(W1, la1, lb1, weff, 512, 512, 512);
    lora_gemm<512, 512, true, false><<<dim3(2, NB / 32, NA), 256, 0, stream>>>(
        hB, weff, b1, hA, 512, nullptr);

    // layer 2: 512 -> 512, relu               hA -> hB
    prep_weff<<<dim3(NA * 512), 256, 0, stream>>>(W2, la2, lb2, weff, 512, 512, 512);
    lora_gemm<512, 512, true, false><<<dim3(2, NB / 32, NA), 256, 0, stream>>>(
        hA, weff, b2, hB, 512, nullptr);

    // layer 3: 512 -> 640, no relu, final     hB -> d_out (f32, own/neigh split)
    prep_weff<<<dim3(NA * 640), 256, 0, stream>>>(W3, la3, lb3, weff, 512, 640, 512);
    lora_gemm<512, 640, false, true><<<dim3(3, NB / 32, NA), 256, 0, stream>>>(
        hB, weff, b3, nullptr, 512, out);
}
